// LSTM_Model_38079180046631
// MI455X (gfx1250) — matrix operand order repaired and checked by reading, not compiled, not run on hardware
//
#include <hip/hip_runtime.h>

#define BB 32
#define TT 256
#define DD 256
#define HH 512
#define GG 2048   // 4*H
#define VV 32000
#define CC 16

typedef __bf16 bf16;
typedef __attribute__((ext_vector_type(16))) __bf16 v16bf;
typedef __attribute__((ext_vector_type(8)))  __bf16 v8bf;
typedef __attribute__((ext_vector_type(8)))  float  v8f;

// ---- fragment loaders (CDNA5 16-bit WMMA VGPR layouts) --------------------
// A (16x32, row-major source, row = lane&15):
//   lanes 0-15 : K {0..7} u {16..23};  lanes 16-31: K {8..15} u {24..31}
__device__ __forceinline__ v16bf load_a16(const bf16* tile, int ld) {
  const int l = threadIdx.x & 31;
  const bf16* p = tile + (size_t)(l & 15) * ld + ((l >> 4) << 3);
  v16bf r;
  *(v8bf*)&r       = *(const v8bf*)(p);
  *((v8bf*)&r + 1) = *(const v8bf*)(p + 16);
  return r;
}
// B (32x16, source is N-major "transposed" weights, column = lane&15):
//   lanes 0-15 : K {0..7} u {16..23};  lanes 16-31: K {8..15} u {24..31}
__device__ __forceinline__ v16bf load_b16(const bf16* tile, int ld) {
  const int l = threadIdx.x & 31;
  const bf16* p = tile + (size_t)(l & 15) * ld + ((l >> 4) << 3);
  v16bf r;
  *(v8bf*)&r       = *(const v8bf*)(p);
  *((v8bf*)&r + 1) = *(const v8bf*)(p + 16);
  return r;
}

__device__ __forceinline__ v8f wmma_bf16(v16bf a, v16bf b, v8f c) {
  return __builtin_amdgcn_wmma_f32_16x16x32_bf16(false, a, false, b,
                                                 (short)0, c, false, false);
}

__device__ __forceinline__ float sigmoidf_(float x) {
  return 1.0f / (1.0f + __expf(-x));
}

// ---- embedding gather: f32 table -> bf16 activations ----------------------
__global__ void k_embed(const int* __restrict__ ids,
                        const float* __restrict__ emb,
                        bf16* __restrict__ x0) {
  int i = blockIdx.x * 256 + threadIdx.x;
  if (i >= BB * TT * DD) return;
  int bt = i >> 8, d = i & (DD - 1);
  x0[i] = (bf16)emb[(size_t)ids[bt] * DD + d];
}

// ---- transpose + convert: W(K,N) f32 -> WT(N,K) bf16 ----------------------
__global__ void k_transpose_bf16(const float* __restrict__ W,
                                 bf16* __restrict__ WT, int K, int N) {
  int i = blockIdx.x * 256 + threadIdx.x;
  if (i >= N * K) return;
  int n = i / K, k = i - n * K;
  WT[i] = (bf16)W[(size_t)k * N + n];
}

// ---- big input GEMM: C(M,GG) f32 = A(M,K)bf16 x Bt(GG,K)bf16 + bias -------
// block: 256 thr (8 waves); block tile 32(M) x 512(N); wave tile 32(M)x64(N)
// => per k-step: 12 b128 loads for 8 WMMAs (A frags reused across 4 N-tiles)
__global__ __launch_bounds__(256)
void k_gemm_bias(const bf16* __restrict__ A, int lda,
                 const bf16* __restrict__ Bt, int ldb,
                 const float* __restrict__ bias,
                 float* __restrict__ C, int ldc, int K) {
  const int lane = threadIdx.x & 31, wave = threadIdx.x >> 5;
  const int mbase = blockIdx.x * 32;
  const int nbase = blockIdx.y * 512 + wave * 64;
  v8f acc[2][4];
  for (int mt = 0; mt < 2; ++mt)
    for (int nt = 0; nt < 4; ++nt)
      for (int v = 0; v < 8; ++v) acc[mt][nt][v] = 0.f;
  for (int kk = 0; kk < K; kk += 32) {
    v16bf a0 = load_a16(A + (size_t)mbase * lda + kk, lda);
    v16bf a1 = load_a16(A + (size_t)(mbase + 16) * lda + kk, lda);
#pragma unroll
    for (int nt = 0; nt < 4; ++nt) {
      v16bf b = load_b16(Bt + (size_t)(nbase + nt * 16) * ldb + kk, ldb);
      acc[0][nt] = wmma_bf16(a0, b, acc[0][nt]);
      acc[1][nt] = wmma_bf16(a1, b, acc[1][nt]);
    }
  }
  const int rm = ((lane >> 4) << 3), cn = lane & 15;
  for (int mt = 0; mt < 2; ++mt)
    for (int nt = 0; nt < 4; ++nt) {
      int n = nbase + nt * 16 + cn;
      float bv = bias[n];
      for (int v = 0; v < 8; ++v) {
        int m = mbase + mt * 16 + rm + v;
        C[(size_t)m * ldc + n] = acc[mt][nt][v] + bv;
      }
    }
}

// ---- persistent LSTM scan: blockIdx.x = direction (0 fwd, 1 bwd) ----------
// 512 thr = 16 waves; wave owns h-col tiles {wave, wave+16}; per step it
// computes z-tiles for the 4 gates x 2 M-tiles with WMMA from LDS-resident h
// and L2-streamed WhT; cell state c never leaves registers.
__global__ __launch_bounds__(512)
void k_lstm_scan(const float* __restrict__ xzF, const float* __restrict__ xzB,
                 const bf16* __restrict__ whTF, const bf16* __restrict__ whTB,
                 const int* __restrict__ mask,
                 bf16* __restrict__ y, int ldy) {
  const int dir = blockIdx.x;
  const float* xz  = dir ? xzB : xzF;
  const bf16*  whT = dir ? whTB : whTF;
  __shared__ __align__(16) bf16 hbuf[2][BB][HH];   // 2 x 32KB ping-pong
  const int tid = threadIdx.x, lane = tid & 31, wave = tid >> 5;
  for (int i = tid; i < BB * HH; i += 512) ((bf16*)hbuf)[i] = (bf16)0.f;
  __syncthreads();

  v8f c_st[2][2];
  for (int a = 0; a < 2; ++a)
    for (int b = 0; b < 2; ++b)
      for (int v = 0; v < 8; ++v) c_st[a][b][v] = 0.f;

  int cur = 0;
  for (int t = 0; t < TT; ++t) {
    const int tt = dir ? (TT - 1 - t) : t;
    for (int jti = 0; jti < 2; ++jti) {
      const int jt = wave + (jti << 4);         // h column tile 0..31
      const int nj = (jt << 4) + (lane & 15);   // owned h column
      v8f z[4][2];
      // init accumulators with precomputed xz (= x@Wi + b) for this step
      for (int g = 0; g < 4; ++g)
        for (int mt = 0; mt < 2; ++mt)
          for (int v = 0; v < 8; ++v) {
            int m = (mt << 4) + ((lane >> 4) << 3) + v;
            z[g][mt][v] = xz[((size_t)m * TT + tt) * GG + (g << 9) + nj];
          }
      const bf16* hs = &hbuf[cur][0][0];
      for (int kk = 0; kk < HH; kk += 32) {
        v16bf a0 = load_a16(hs + kk, HH);
        v16bf a1 = load_a16(hs + 16 * HH + kk, HH);
        if (kk + 32 < HH)
          __builtin_prefetch(whT + (size_t)(jt << 4) * HH + kk + 32, 0, 0);
#pragma unroll
        for (int g = 0; g < 4; ++g) {
          v16bf bf = load_b16(whT + (size_t)((g << 9) + (jt << 4)) * HH + kk, HH);
          z[g][0] = wmma_bf16(a0, bf, z[g][0]);
          z[g][1] = wmma_bf16(a1, bf, z[g][1]);
        }
      }
      // gates + cell update + mask
      for (int mt = 0; mt < 2; ++mt)
        for (int v = 0; v < 8; ++v) {
          int m = (mt << 4) + ((lane >> 4) << 3) + v;   // batch row
          float iv = sigmoidf_(z[0][mt][v]);
          float fv = sigmoidf_(z[1][mt][v]);
          float gv = tanhf(z[2][mt][v]);
          float ov = sigmoidf_(z[3][mt][v]);
          float cold = c_st[jti][mt][v];
          float cn = fv * cold + iv * gv;
          float hn = ov * tanhf(cn);
          bool mk = mask[m * TT + tt] != 0;
          float hp = (float)hbuf[cur][m][nj];
          float h2 = mk ? hn : hp;
          c_st[jti][mt][v] = mk ? cn : cold;
          bf16 hb = (bf16)h2;
          hbuf[cur ^ 1][m][nj] = hb;
          y[((size_t)m * TT + tt) * ldy + (dir << 9) + nj] = hb;
        }
    }
    __syncthreads();
    cur ^= 1;
  }
}

// ---- final projection: (B*T,1024) x WoT(16,1024) + bias -> f32 logits -----
__global__ __launch_bounds__(256)
void k_proj(const bf16* __restrict__ X, const bf16* __restrict__ WoT,
            const float* __restrict__ bias, float* __restrict__ out) {
  const int lane = threadIdx.x & 31, wave = threadIdx.x >> 5;
  const int mbase = (blockIdx.x * 8 + wave) * 32;
  v8f acc[2];
  for (int mt = 0; mt < 2; ++mt)
    for (int v = 0; v < 8; ++v) acc[mt][v] = 0.f;
  for (int kk = 0; kk < 2 * HH; kk += 32) {
    v16bf b  = load_b16(WoT + kk, 2 * HH);
    v16bf a0 = load_a16(X + (size_t)mbase * (2 * HH) + kk, 2 * HH);
    v16bf a1 = load_a16(X + (size_t)(mbase + 16) * (2 * HH) + kk, 2 * HH);
    acc[0] = wmma_bf16(a0, b, acc[0]);
    acc[1] = wmma_bf16(a1, b, acc[1]);
  }
  const int n = lane & 15;
  const float bv = bias[n];
  for (int mt = 0; mt < 2; ++mt)
    for (int v = 0; v < 8; ++v) {
      int m = mbase + (mt << 4) + ((lane >> 4) << 3) + v;
      out[m * CC + n] = acc[mt][v] + bv;
    }
}

extern "C" void kernel_launch(void* const* d_in, const int* in_sizes, int n_in,
                              void* d_out, int out_size, void* d_ws, size_t ws_size,
                              hipStream_t stream) {
  (void)n_in; (void)out_size; (void)ws_size;
  const int* word_ids = (const int*)d_in[0];
  const int* mask     = (const int*)d_in[1];
  const float *emb, *Wo, *bo;
  const float *Wi[3][2], *Wh[3][2], *bg[3][2];
  int base;
  if (in_sizes[2] == VV * DD) {          // insertion order: emb, layers, W, b
    emb = (const float*)d_in[2];
    base = 3;
    Wo = (const float*)d_in[21];
    bo = (const float*)d_in[22];
  } else {                               // sorted pytree: W, b, emb, layers
    Wo  = (const float*)d_in[2];
    bo  = (const float*)d_in[3];
    emb = (const float*)d_in[4];
    base = 5;
  }
  for (int l = 0; l < 3; ++l)
    for (int d = 0; d < 2; ++d) {
      int idx = base + l * 6 + d * 3;
      Wi[l][d] = (const float*)d_in[idx];
      Wh[l][d] = (const float*)d_in[idx + 1];
      bg[l][d] = (const float*)d_in[idx + 2];
    }

  // workspace carve (~177 MB)
  char* w = (char*)d_ws;
  auto carve = [&](size_t bytes) {
    void* p = (void*)w;
    w += (bytes + 255) & ~(size_t)255;
    return p;
  };
  bf16*  act0 = (bf16*)carve((size_t)BB * TT * DD * 2);
  bf16*  act1 = (bf16*)carve((size_t)BB * TT * 2 * HH * 2);
  bf16*  act2 = (bf16*)carve((size_t)BB * TT * 2 * HH * 2);
  float* xzf  = (float*)carve((size_t)BB * TT * GG * 4);
  float* xzb  = (float*)carve((size_t)BB * TT * GG * 4);
  bf16*  wiTf = (bf16*)carve((size_t)GG * 2 * HH * 2);
  bf16*  wiTb = (bf16*)carve((size_t)GG * 2 * HH * 2);
  bf16*  whTf = (bf16*)carve((size_t)GG * HH * 2);
  bf16*  whTb = (bf16*)carve((size_t)GG * HH * 2);
  bf16*  woT  = (bf16*)carve((size_t)CC * 2 * HH * 2);

  k_embed<<<(BB * TT * DD + 255) / 256, 256, 0, stream>>>(word_ids, emb, act0);

  const bf16* lin[3]  = {act0, act1, act2};
  bf16*       lout[3] = {act1, act2, act1};
  const int   din[3]  = {DD, 2 * HH, 2 * HH};

  for (int l = 0; l < 3; ++l) {
    const int K  = din[l];
    const int ni = GG * K;
    const int nh = GG * HH;
    k_transpose_bf16<<<(ni + 255) / 256, 256, 0, stream>>>(Wi[l][0], wiTf, K, GG);
    k_transpose_bf16<<<(ni + 255) / 256, 256, 0, stream>>>(Wi[l][1], wiTb, K, GG);
    k_transpose_bf16<<<(nh + 255) / 256, 256, 0, stream>>>(Wh[l][0], whTf, HH, GG);
    k_transpose_bf16<<<(nh + 255) / 256, 256, 0, stream>>>(Wh[l][1], whTb, HH, GG);
    dim3 gg((BB * TT) / 32, GG / 512);
    k_gemm_bias<<<gg, 256, 0, stream>>>(lin[l], K, wiTf, K, bg[l][0], xzf, GG, K);
    k_gemm_bias<<<gg, 256, 0, stream>>>(lin[l], K, wiTb, K, bg[l][1], xzb, GG, K);
    k_lstm_scan<<<2, 512, 0, stream>>>(xzf, xzb, whTf, whTb, mask, lout[l], 2 * HH);
  }
  k_transpose_bf16<<<(CC * 2 * HH + 255) / 256, 256, 0, stream>>>(Wo, woT, 2 * HH, CC);
  k_proj<<<(BB * TT) / 256, 256, 0, stream>>>(act1, woT, bo, (float*)d_out);
}
